// DGLFeatureGAT_23922967839174
// MI455X (gfx1250) — matrix operand order repaired and checked by reading, not compiled, not run on hardware
//
#include <hip/hip_runtime.h>

// GATv2 (dense complete graph) fused pipeline for gfx1250 / MI455X.
//   prep kernel : one-time fp32->bf16 convert + transpose of Wl, Wr, x into d_ws
//   main kernel : one WG per batch (32 WGs x 256 thr); async-to-LDS staging,
//                 bf16 WMMA (f32 accum) GEMMs, fp32 VALU logits+softmax.
// Fallback (small ws): round-1 style fully self-contained kernel.

typedef __attribute__((ext_vector_type(16))) __bf16 v16bf;
typedef __attribute__((ext_vector_type(8)))  __bf16 v8bf;
typedef __attribute__((ext_vector_type(8)))  float  v8f;

#define B_ 32
#define W_ 128   // node feature dim (GEMM K)
#define F_ 64    // number of nodes
#define H_ 2
#define D_ 128
#define HD 256
#define NEG_SLOPE 0.2f

// ================= CDNA5 async global->LDS copy (ASYNCcnt path) =============
__device__ __forceinline__ void async_copy16(void* lds_dst, const void* gsrc) {
  // LDS address is the low 32 bits of the shared-memory pointer.
  uint32_t l = (uint32_t)(uintptr_t)lds_dst;
  asm volatile("global_load_async_to_lds_b128 %0, %1, off"
               :: "v"(l), "v"(gsrc) : "memory");
}
__device__ __forceinline__ void wait_async0() {
  asm volatile("s_wait_asynccnt 0x0" ::: "memory");
}

// ===================== WMMA fragment helpers (wave32) ========================
// A (16x32 bf16 MxK) from row-major [M][ld]:
// lanes 0-15: M=lane, K=[0,16); lanes 16-31: M=lane-16, K=[16,32)
__device__ __forceinline__ v16bf load_a(const __bf16* base, int ld, int m0, int k0, int lane) {
  int m   = m0 + (lane & 15);
  int klo = k0 + ((lane >> 4) << 4);
  const __bf16* p = base + m * ld + klo;
  v8bf lo = *(const v8bf*)p;
  v8bf hi = *(const v8bf*)(p + 8);
  v16bf r;
#pragma unroll
  for (int i = 0; i < 8; ++i) { r[i] = lo[i]; r[i + 8] = hi[i]; }
  return r;
}
// B (32x16 bf16 KxN) from TRANSPOSED [N][ld] storage: lane N=lane%16,
// K = 16*(lane/16)+[0,16) -> one contiguous 32B load.
__device__ __forceinline__ v16bf load_b(const __bf16* baseT, int ld, int n0, int k0, int lane) {
  int n  = n0 + (lane & 15);
  int kb = k0 + ((lane >> 4) << 4);
  return *(const v16bf*)(baseT + n * ld + kb);
}
// Store 16x16 f32 D transposed as bf16 into [N][ldt].
__device__ __forceinline__ void store_d_T_bf16(__bf16* dT, int ldt, int m0, int n0, int lane, v8f acc) {
  int n  = n0 + (lane & 15);
  int mb = m0 + ((lane >> 4) << 3);
#pragma unroll
  for (int r = 0; r < 8; ++r) dT[n * ldt + mb + r] = (__bf16)acc[r];
}

// ==================== shared phase 2/3 (attention + output) ==================
__device__ __forceinline__ void attention_phase(const __bf16* ELT, const __bf16* ERT,
                                                const float* ATT, float* ESC,
                                                __bf16* ALPHA, int tid) {
  { // logits e[h,i,j] = a_h . lrelu(el_j + er_i); 2 threads per (h,i) row
    int pair = tid & 127, half = tid >> 7;
    int h = pair >> 6, i = pair & 63;
    const __bf16* elh = ELT + h * D_ * F_;
    const __bf16* erh = ERT + h * D_ * F_;
    const float*  ah  = ATT + h * D_;
    for (int j = half * 32; j < half * 32 + 32; ++j) {
      float acc = 0.f;
#pragma unroll 4
      for (int d = 0; d < D_; ++d) {
        float v = (float)elh[d * F_ + j] + (float)erh[d * F_ + i];
        v = (v > 0.f) ? v : NEG_SLOPE * v;
        acc += ah[d] * v;
      }
      ESC[pair * F_ + j] = acc;
    }
  }
  __syncthreads();
  if (tid < 128) { // softmax over j
    float m = -3.4e38f;
    for (int j = 0; j < F_; ++j) m = fmaxf(m, ESC[tid * F_ + j]);
    float ssum = 0.f;
    for (int j = 0; j < F_; ++j) ssum += __expf(ESC[tid * F_ + j] - m);
    float inv = 1.f / ssum;
    for (int j = 0; j < F_; ++j)
      ALPHA[tid * F_ + j] = (__bf16)(__expf(ESC[tid * F_ + j] - m) * inv);
  }
  __syncthreads();
}

__device__ __forceinline__ void output_phase(const __bf16* ELT, const __bf16* ALPHA,
                                             const float* BM, float* out,
                                             int b, int wave, int lane) {
  for (int t = wave; t < 32; t += 8) {            // 4 i-tiles x 8 d-tiles
    int mt = t >> 3, nt = t & 7;
    v8f acc = {};                                 // accumulates BOTH heads
#pragma unroll
    for (int h = 0; h < 2; ++h) {
      const __bf16* al  = ALPHA + h * F_ * F_;    // [i][j] row-major
      const __bf16* elh = ELT + h * D_ * F_;      // [d][j] = B^T layout
#pragma unroll
      for (int kt = 0; kt < 2; ++kt) {            // K = 64 nodes
        v16bf af = load_a(al, F_, mt * 16, kt * 32, lane);
        v16bf bf = load_b(elh, F_, nt * 16, kt * 32, lane);
        acc = __builtin_amdgcn_wmma_f32_16x16x32_bf16(false, af, false, bf,
                                                      (short)0, acc, false, false);
      }
    }
    int d  = nt * 16 + (lane & 15);
    int ib = mt * 16 + ((lane >> 4) << 3);
    float bm = BM[d];
    float* op = out + b * (D_ * F_) + d * F_ + ib;  // 8 contiguous f32 per lane
#pragma unroll
    for (int r = 0; r < 8; ++r) op[r] = 0.5f * acc[r] + bm;
  }
}

// ========================== prep kernel (one-time) ===========================
__global__ void gatv2_prep_kernel(const float* __restrict__ x,
                                  const float* __restrict__ Wl,
                                  const float* __restrict__ Wr,
                                  __bf16* __restrict__ WLTg,   // [256][128]
                                  __bf16* __restrict__ WRTg,   // [256][128]
                                  __bf16* __restrict__ XTall) {// [B][64][128]
  int idx = blockIdx.x * 256 + threadIdx.x;
  int stride = gridDim.x * 256;
  for (int i = idx; i < W_ * HD; i += stride) {     // WT[o][w] = W[w][o]
    int o = i >> 7, w = i & 127;                    // writes coalesced over w
    WLTg[i] = (__bf16)Wl[w * HD + o];
    WRTg[i] = (__bf16)Wr[w * HD + o];
  }
  for (int i = idx; i < B_ * F_ * W_; i += stride) { // XT[b][n][w] = x[b][w][n]
    int bb = i >> 13, r = i & 8191;
    int n = r >> 7, w = r & 127;                     // writes coalesced over w
    XTall[i] = (__bf16)x[bb * (W_ * F_) + w * F_ + n];
  }
}

// ============================ main fused kernel ==============================
// LDS layout (bytes):
#define M_XT    0        // bf16 [64][128]   16384
#define M_WLT   16384    // bf16 [256][128]  65536
#define M_WRT   81920    // bf16 [256][128]  65536
#define M_ELT   147456   // bf16 [256][64]   32768
#define M_ERT   180224   // bf16 [256][64]   32768
#define M_ATT   212992   // f32  [256]        1024
#define M_BM    214016   // f32  [128]         512
#define M_BYTES 214528
#define M_ESC   16384    // alias in WLT: f32 [128][64]
#define M_ALPHA 49152    // alias in WLT: bf16 [128][64]

__global__ void gatv2_main_kernel(const __bf16* __restrict__ XTall,
                                  const __bf16* __restrict__ WLTg,
                                  const __bf16* __restrict__ WRTg,
                                  const float* __restrict__ attn_a,
                                  const float* __restrict__ bias,
                                  float* __restrict__ out) {
  extern __shared__ char smem[];
  __bf16* XT    = (__bf16*)(smem + M_XT);
  __bf16* WLT   = (__bf16*)(smem + M_WLT);
  __bf16* WRT   = (__bf16*)(smem + M_WRT);
  __bf16* ELT   = (__bf16*)(smem + M_ELT);
  __bf16* ERT   = (__bf16*)(smem + M_ERT);
  float*  ATT   = (float*)(smem + M_ATT);
  float*  BM    = (float*)(smem + M_BM);
  float*  ESC   = (float*)(smem + M_ESC);
  __bf16* ALPHA = (__bf16*)(smem + M_ALPHA);

  const int b = blockIdx.x, tid = threadIdx.x;
  const int lane = tid & 31, wave = tid >> 5;

  // ---- Phase 0: async-stage bf16 tiles straight into LDS (ASYNCcnt) ----
  const char* xsrc = (const char*)(XTall + b * (F_ * W_));
  for (int c = tid; c < 1024; c += 256)          // XT: 16 KB
    async_copy16((char*)XT + c * 16, xsrc + c * 16);
  for (int c = tid; c < 4096; c += 256)          // WLT: 64 KB
    async_copy16((char*)WLT + c * 16, (const char*)WLTg + c * 16);
  for (int c = tid; c < 4096; c += 256)          // WRT: 64 KB
    async_copy16((char*)WRT + c * 16, (const char*)WRTg + c * 16);
  for (int i = tid; i < HD; i += 256) ATT[i] = attn_a[i];
  for (int i = tid; i < D_; i += 256) BM[i] = 0.5f * (bias[i] + bias[D_ + i]);
  wait_async0();
  __syncthreads();

  // ---- Phase 1: el, er in ONE pass (128 tiles of 16x16, K=128) ----
  for (int t = wave; t < 128; t += 8) {
    int s = t >> 6, tt = t & 63;
    int mt = tt >> 4, nt = tt & 15;
    const __bf16* WT = s ? WRT : WLT;
    __bf16* OUT = s ? ERT : ELT;
    v8f acc = {};
#pragma unroll
    for (int kt = 0; kt < 4; ++kt) {
      v16bf af = load_a(XT, W_, mt * 16, kt * 32, lane);
      v16bf bf = load_b(WT, W_, nt * 16, kt * 32, lane);
      acc = __builtin_amdgcn_wmma_f32_16x16x32_bf16(false, af, false, bf,
                                                    (short)0, acc, false, false);
    }
    store_d_T_bf16(OUT, F_, mt * 16, nt * 16, lane, acc);   // [o][node]
  }
  __syncthreads();

  attention_phase(ELT, ERT, ATT, ESC, ALPHA, tid);
  output_phase(ELT, ALPHA, BM, out, b, wave, lane);
}

// ================= fallback: self-contained fused kernel (round 1) ===========
#define S_XT    0
#define S_WT    16384
#define S_ELT   81920
#define S_ERT   114688
#define S_ATT   147456
#define S_BM    148480
#define S_BYTES 148992
#define S_ESC   16384
#define S_ALPHA 49152

__global__ void gatv2_fused_kernel(const float* __restrict__ x,
                                   const float* __restrict__ Wl,
                                   const float* __restrict__ Wr,
                                   const float* __restrict__ attn_a,
                                   const float* __restrict__ bias,
                                   float* __restrict__ out) {
  extern __shared__ char smem[];
  __bf16* XT    = (__bf16*)(smem + S_XT);
  __bf16* WT    = (__bf16*)(smem + S_WT);
  __bf16* ELT   = (__bf16*)(smem + S_ELT);
  __bf16* ERT   = (__bf16*)(smem + S_ERT);
  float*  ATT   = (float*)(smem + S_ATT);
  float*  BM    = (float*)(smem + S_BM);
  float*  ESC   = (float*)(smem + S_ESC);
  __bf16* ALPHA = (__bf16*)(smem + S_ALPHA);

  const int b = blockIdx.x, tid = threadIdx.x;
  const int lane = tid & 31, wave = tid >> 5;
  const float* xb = x + b * (W_ * F_);

  for (int idx = tid; idx < F_ * W_; idx += 256) {
    int w = idx >> 6, n = idx & 63;
    XT[n * W_ + w] = (__bf16)xb[w * F_ + n];
  }
  for (int idx = tid; idx < HD; idx += 256) ATT[idx] = attn_a[idx];
  for (int idx = tid; idx < D_; idx += 256) BM[idx] = 0.5f * (bias[idx] + bias[D_ + idx]);

  for (int s = 0; s < 2; ++s) {
    const float* Wg = (s == 0) ? Wl : Wr;
    __bf16* OUT = (s == 0) ? ELT : ERT;
    __syncthreads();
    for (int idx = tid; idx < W_ * HD; idx += 256) {
      int w = idx >> 8, o = idx & 255;
      WT[o * W_ + w] = (__bf16)Wg[w * HD + o];
    }
    __syncthreads();
    for (int t = wave; t < 64; t += 8) {
      int mt = t >> 4, nt = t & 15;
      v8f acc = {};
#pragma unroll
      for (int kt = 0; kt < 4; ++kt) {
        v16bf af = load_a(XT, W_, mt * 16, kt * 32, lane);
        v16bf bf = load_b(WT, W_, nt * 16, kt * 32, lane);
        acc = __builtin_amdgcn_wmma_f32_16x16x32_bf16(false, af, false, bf,
                                                      (short)0, acc, false, false);
      }
      store_d_T_bf16(OUT, F_, mt * 16, nt * 16, lane, acc);
    }
  }
  __syncthreads();

  attention_phase(ELT, ERT, ATT, ESC, ALPHA, tid);
  output_phase(ELT, ALPHA, BM, out, b, wave, lane);
}

// ================================ launcher ===================================
extern "C" void kernel_launch(void* const* d_in, const int* in_sizes, int n_in,
                              void* d_out, int out_size, void* d_ws, size_t ws_size,
                              hipStream_t stream) {
  const float* x      = (const float*)d_in[0];
  const float* Wl     = (const float*)d_in[1];
  const float* Wr     = (const float*)d_in[2];
  const float* attn_a = (const float*)d_in[3];
  const float* bias   = (const float*)d_in[4];
  float* out = (float*)d_out;

  const size_t need = 2u * 65536u + (size_t)B_ * F_ * W_ * 2u;  // 655360 B
  if (d_ws != nullptr && ws_size >= need) {
    __bf16* WLTg  = (__bf16*)d_ws;
    __bf16* WRTg  = (__bf16*)((char*)d_ws + 65536);
    __bf16* XTall = (__bf16*)((char*)d_ws + 131072);
    gatv2_prep_kernel<<<dim3(128), dim3(256), 0, stream>>>(x, Wl, Wr, WLTg, WRTg, XTall);
    (void)hipFuncSetAttribute((const void*)gatv2_main_kernel,
                              hipFuncAttributeMaxDynamicSharedMemorySize, M_BYTES);
    gatv2_main_kernel<<<dim3(B_), dim3(256), M_BYTES, stream>>>(XTall, WLTg, WRTg,
                                                                attn_a, bias, out);
  } else {
    (void)hipFuncSetAttribute((const void*)gatv2_fused_kernel,
                              hipFuncAttributeMaxDynamicSharedMemorySize, S_BYTES);
    gatv2_fused_kernel<<<dim3(B_), dim3(256), S_BYTES, stream>>>(x, Wl, Wr,
                                                                 attn_a, bias, out);
  }
}
